// RNNLayerForPC_2491081031813
// MI455X (gfx1250) — matrix operand order repaired and checked by reading, not compiled, not run on hardware
//
#include <hip/hip_runtime.h>
#include <hip/hip_bf16.h>
#include <math.h>

// Problem dims (fixed per reference)
#define S_LEN 512
#define BATCH 128
#define I_DIM 512
#define H_DIM 512
#define BH    (BATCH * H_DIM)          // 65536 elements per timestep
#define SBH   (S_LEN * BH)             // 33554432 elements per output tensor

// Persistent scan kernel config
#define NBLK          32
#define SCAN_THREADS  256
#define SCAN_GTHREADS (NBLK * SCAN_THREADS)   // 8192

typedef __attribute__((ext_vector_type(16))) _Float16 v16h;
typedef __attribute__((ext_vector_type(8)))  _Float16 v8h;
typedef __attribute__((ext_vector_type(8)))  float    v8f;

// ---------------------------------------------------------------------------
// WMMA fragment helpers (CDNA5 16x16x32 f16 layouts, ISA 7.12.2, wave32)
// ---------------------------------------------------------------------------

// A fragment (16x32, row-major A[M][K], lda in elements) from f16 source.
// Lane L holds row m0+(L&15). Lanes 0-15: K = k0+{0..7, 16..23};
// lanes 16-31: +8 on both runs. Two contiguous 16B loads.
__device__ inline v16h load_a_f16(const _Float16* __restrict__ A, int lda,
                                  int m0, int k0, int lane) {
  const _Float16* p = A + (size_t)(m0 + (lane & 15)) * lda + k0 + ((lane & 16) ? 8 : 0);
  v8h lo = *(const v8h*)(p);
  v8h hi = *(const v8h*)(p + 16);
  return __builtin_shufflevector(lo, hi, 0,1,2,3,4,5,6,7,8,9,10,11,12,13,14,15);
}

// A fragment from f32 source with on-the-fly f32->f16 convert (same K mapping).
__device__ inline v16h load_a_f32cvt(const float* __restrict__ A, int lda,
                                     int m0, int k0, int lane) {
  const float* p = A + (size_t)(m0 + (lane & 15)) * lda + k0 + ((lane & 16) ? 8 : 0);
  v16h r;
#pragma unroll
  for (int j = 0; j < 8; ++j) {
    r[j]     = (_Float16)p[j];
    r[j + 8] = (_Float16)p[16 + j];
  }
  return r;
}

// B fragment (32x16) where B[k][n] = W[n][k], W row-major [N][K] (nn.Linear
// weight layout). Lane L holds column n0+(L&15); lanes 0-15 hold K=k0+{0..7, 16..23},
// lanes 16-31: +8 on both runs. Two contiguous 16B loads per lane.
__device__ inline v16h load_b_f16(const _Float16* __restrict__ W, int ldw,
                                  int n0, int k0, int lane) {
  const _Float16* p = W + (size_t)(n0 + (lane & 15)) * ldw + k0 + ((lane & 16) ? 8 : 0);
  v8h lo = *(const v8h*)(p);
  v8h hi = *(const v8h*)(p + 16);
  return __builtin_shufflevector(lo, hi, 0,1,2,3,4,5,6,7,8,9,10,11,12,13,14,15);
}

// C/D 16x16 f32: VGPR v holds row m0+v (lanes 0-15) or m0+8+v (lanes 16-31),
// column n0+(L&15).
__device__ inline v8f load_c_f32(const float* __restrict__ C, int ldc,
                                 int m0, int n0, int lane) {
  const int n  = n0 + (lane & 15);
  const int mb = m0 + ((lane & 16) ? 8 : 0);
  v8f c;
#pragma unroll
  for (int v = 0; v < 8; ++v) c[v] = C[(size_t)(mb + v) * ldc + n];
  return c;
}

__device__ inline void store_d_f32(float* __restrict__ D, int ldd,
                                   int m0, int n0, int lane, v8f acc) {
  const int n  = n0 + (lane & 15);
  const int mb = m0 + ((lane & 16) ? 8 : 0);
#pragma unroll
  for (int v = 0; v < 8; ++v) D[(size_t)(mb + v) * ldd + n] = acc[v];
}

__device__ inline v8f wmma_f16(v16h a, v16h b, v8f c) {
  // (neg_a, A, neg_b, B, c_mod, C, reuse_a, reuse_b)
  return __builtin_amdgcn_wmma_f32_16x16x32_f16(false, a, false, b, (short)0, c,
                                                false, false);
}

// ---------------------------------------------------------------------------
// Small utility kernels
// ---------------------------------------------------------------------------

__global__ void cvt_f32_f16_kernel(const float* __restrict__ src,
                                   _Float16* __restrict__ dst, int n) {
  int i = blockIdx.x * blockDim.x + threadIdx.x;
  if (i < n) dst[i] = (_Float16)src[i];
}

__global__ void init_bar_kernel(unsigned* bar) {
  bar[0] = 0u;  // arrival count
  bar[1] = 0u;  // generation
}

// ---------------------------------------------------------------------------
// Kernel 1: xp = x @ W_in^T   (M=S*B=65536, N=H=512, K=I=512)
// Each wave computes a 2x2 block of 16x16 tiles (32x32 output region).
// x is converted f32->f16 per fragment (x fits in the 192MB L2, reuse is cheap).
// Result written straight into the `prediction` slab of d_out.
// ---------------------------------------------------------------------------
__global__ void __launch_bounds__(256)
xp_gemm_kernel(const float* __restrict__ x, const _Float16* __restrict__ Win16,
               float* __restrict__ out) {
  const int lane = threadIdx.x & 31;
  const int wid  = threadIdx.x >> 5;
  const int wg   = blockIdx.x * 8 + wid;   // 0 .. 32767
  const int rm   = wg >> 4;                // 0 .. 2047 (M regions of 32)
  const int rn   = wg & 15;                // 0 .. 15   (N regions of 32)
  const int m0   = rm * 32;
  const int n0   = rn * 32;

  v8f a00 = {}, a01 = {}, a10 = {}, a11 = {};
#pragma unroll 4
  for (int k0 = 0; k0 < I_DIM; k0 += 32) {
    v16h fa0 = load_a_f32cvt(x, I_DIM, m0,      k0, lane);
    v16h fa1 = load_a_f32cvt(x, I_DIM, m0 + 16, k0, lane);
    v16h fb0 = load_b_f16(Win16, I_DIM, n0,      k0, lane);
    v16h fb1 = load_b_f16(Win16, I_DIM, n0 + 16, k0, lane);
    a00 = wmma_f16(fa0, fb0, a00);
    a01 = wmma_f16(fa0, fb1, a01);
    a10 = wmma_f16(fa1, fb0, a10);
    a11 = wmma_f16(fa1, fb1, a11);
  }
  store_d_f32(out, H_DIM, m0,      n0,      lane, a00);
  store_d_f32(out, H_DIM, m0,      n0 + 16, lane, a01);
  store_d_f32(out, H_DIM, m0 + 16, n0,      lane, a10);
  store_d_f32(out, H_DIM, m0 + 16, n0 + 16, lane, a11);
}

// ---------------------------------------------------------------------------
// Grid-wide sense barrier (bar[0]=count, bar[1]=generation), NBLK blocks.
// ---------------------------------------------------------------------------
__device__ inline void grid_sync(unsigned* bar, unsigned* gen_local) {
  __threadfence();
  __syncthreads();
  if (threadIdx.x == 0) {
    const unsigned target = *gen_local + 1;
    unsigned arrived = __hip_atomic_fetch_add(&bar[0], 1u, __ATOMIC_ACQ_REL,
                                              __HIP_MEMORY_SCOPE_AGENT);
    if (arrived == NBLK - 1) {
      __hip_atomic_store(&bar[0], 0u, __ATOMIC_RELAXED, __HIP_MEMORY_SCOPE_AGENT);
      __hip_atomic_store(&bar[1], target, __ATOMIC_RELEASE, __HIP_MEMORY_SCOPE_AGENT);
    } else {
      while (__hip_atomic_load(&bar[1], __ATOMIC_ACQUIRE,
                               __HIP_MEMORY_SCOPE_AGENT) < target) {
        __builtin_amdgcn_s_sleep(2);
      }
    }
  }
  __syncthreads();
  *gen_local += 1;
}

// ---------------------------------------------------------------------------
// Kernel 2: persistent recurrence.
//   pred[t] currently holds xp[t]; for t=1..S-1:
//     phase A: th = tanh(pred[t-1])  -> tanh_out[t-1] (f32) and th16 (f16)
//     phase B: pred[t] = th16 @ W_h^T + pred[t]   (256 waves = 8x32 tiles)
//   finally tanh_out[S-1] = tanh(pred[S-1]).
// KEY: each wave's 16 B-fragments of W_h (its 16-column strip, all K) are
// hoisted into 128 VGPRs before the time loop — W_h is loaded exactly once
// per wave for all 511 steps, halving the loads on the serial critical path.
// ---------------------------------------------------------------------------
__global__ void __launch_bounds__(SCAN_THREADS)
scan_kernel(float* __restrict__ pred, float* __restrict__ tanh_out,
            _Float16* __restrict__ th16, const _Float16* __restrict__ Wh16,
            unsigned* bar) {
  const int tid     = threadIdx.x;
  const int lane    = tid & 31;
  const int wid     = tid >> 5;
  const int gthread = blockIdx.x * SCAN_THREADS + tid;  // 0..8191
  const int gwave   = blockIdx.x * 8 + wid;             // 0..255
  const int m0      = (gwave >> 5) * 16;                // 8 tile rows  (B=128)
  const int n0      = (gwave & 31) * 16;                // 32 tile cols (H=512)
  unsigned gen = 0;

  // Register-resident W_h fragments: 16 x v16h = 128 VGPRs (occupancy is
  // irrelevant: only 32 persistent blocks on the whole device).
  v16h wb[H_DIM / 32];
#pragma unroll
  for (int kk = 0; kk < H_DIM / 32; ++kk)
    wb[kk] = load_b_f16(Wh16, H_DIM, n0, kk * 32, lane);

  for (int t = 1; t < S_LEN; ++t) {
    // ---- phase A: tanh of previous hidden state (computed exactly once) ----
    const float* hp = pred + (size_t)(t - 1) * BH;
    float*       to = tanh_out + (size_t)(t - 1) * BH;
#pragma unroll
    for (int r = 0; r < BH / SCAN_GTHREADS; ++r) {   // 8 elements / thread
      const int idx = gthread + r * SCAN_GTHREADS;
      const float v = tanhf(hp[idx]);
      to[idx]   = v;
      th16[idx] = (_Float16)v;
    }
    grid_sync(bar, &gen);

    // ---- phase B: one 16x16 WMMA tile per wave, K=512 fully unrolled -------
    float* out = pred + (size_t)t * BH;              // holds xp[t] on entry
    v8f acc = load_c_f32(out, H_DIM, m0, n0, lane);  // residual add init
#pragma unroll
    for (int kk = 0; kk < H_DIM / 32; ++kk) {
      v16h fa = load_a_f16(th16, H_DIM, m0, kk * 32, lane);
      acc = wmma_f16(fa, wb[kk], acc);
    }
    store_d_f32(out, H_DIM, m0, n0, lane, acc);
    grid_sync(bar, &gen);
  }

  // ---- tail: tanh of the last hidden state --------------------------------
  {
    const float* hp = pred + (size_t)(S_LEN - 1) * BH;
    float*       to = tanh_out + (size_t)(S_LEN - 1) * BH;
#pragma unroll
    for (int r = 0; r < BH / SCAN_GTHREADS; ++r) {
      const int idx = gthread + r * SCAN_GTHREADS;
      to[idx] = tanhf(hp[idx]);
    }
  }
}

// ---------------------------------------------------------------------------
// Host-side launch
// ---------------------------------------------------------------------------
extern "C" void kernel_launch(void* const* d_in, const int* in_sizes, int n_in,
                              void* d_out, int out_size, void* d_ws, size_t ws_size,
                              hipStream_t stream) {
  (void)in_sizes; (void)n_in; (void)out_size; (void)ws_size;

  const float* x    = (const float*)d_in[0];   // [S,B,I]
  const float* W_in = (const float*)d_in[1];   // [H,I]
  const float* W_h  = (const float*)d_in[2];   // [H,H]

  float* pred = (float*)d_out;                 // [S,B,H] prediction
  float* tout = pred + (size_t)SBH;            // [S,B,H] tanh(prediction)

  // Workspace layout (~1.2 MB total):
  char* ws = (char*)d_ws;
  _Float16* Wh16  = (_Float16*)(ws);                        // 512 KB
  _Float16* Win16 = (_Float16*)(ws + 512 * 1024);           // 512 KB
  _Float16* th16  = (_Float16*)(ws + 1024 * 1024);          // 128 KB
  unsigned* bar   = (unsigned*)(ws + 1024 * 1024 + 256 * 1024);

  // 1) f32 -> f16 weight copies (L2-resident afterwards)
  cvt_f32_f16_kernel<<<(H_DIM * I_DIM) / 256, 256, 0, stream>>>(W_in, Win16,
                                                                H_DIM * I_DIM);
  cvt_f32_f16_kernel<<<(H_DIM * H_DIM) / 256, 256, 0, stream>>>(W_h, Wh16,
                                                                H_DIM * H_DIM);
  init_bar_kernel<<<1, 1, 0, stream>>>(bar);

  // 2) Bulk input projection: xp straight into the prediction slab.
  //    65536x512 output / (32x32 per wave) = 32768 waves = 4096 blocks.
  xp_gemm_kernel<<<4096, 256, 0, stream>>>(x, Win16, pred);

  // 3) Persistent serial scan (fixed 32 blocks; barrier assumes gridDim==NBLK).
  scan_kernel<<<NBLK, SCAN_THREADS, 0, stream>>>(pred, tout, th16, Wh16, bar);
}
